// Attention_45097156608219
// MI455X (gfx1250) — matrix operand order repaired and checked by reading, not compiled, not run on hardware
//
#include <hip/hip_runtime.h>

typedef __attribute__((ext_vector_type(16))) __bf16 v16bf;
typedef __attribute__((ext_vector_type(4)))  __bf16 v4bf;
typedef __attribute__((ext_vector_type(8)))  float  v8f;

#define B_   32
#define S_   2048
#define H_   1024
#define U_   1024
#define TILE_M 128

// ---------------------------------------------------------------------------
// Kernel 1: W1 (fp32 [H,U]) -> bf16, pre-swizzled into WMMA B-fragment order.
// Fragment addressing: frag(kb, nt) lane L elem j  ->  K = kb*32 + (L<16?0:16)+j,
// N = nt*16 + (L&15).  Stored contiguously: Wb[((kb*64+nt)*32 + L)*16 + j].
// ---------------------------------------------------------------------------
__global__ void swizzle_w1(const float* __restrict__ W1, __bf16* __restrict__ Wb) {
    int idx  = blockIdx.x * 256 + threadIdx.x;      // 0..65535
    int lane = idx & 31;
    int nt   = (idx >> 5) & 63;
    int kb   = idx >> 11;                           // 0..31
    int n     = nt * 16 + (lane & 15);
    int kbase = kb * 32 + ((lane < 16) ? 0 : 16);
    __bf16* dst = Wb + (size_t)idx * 16;
#pragma unroll
    for (int j = 0; j < 16; ++j)
        dst[j] = (__bf16)W1[(size_t)(kbase + j) * U_ + n];
}

// ---------------------------------------------------------------------------
// Kernel 2: qc[b][u] = query[b] @ W2[:,u] + b2[u] + b1[u]   (tiny GEMV)
// ---------------------------------------------------------------------------
__global__ void qproj(const float* __restrict__ q, const float* __restrict__ W2,
                      const float* __restrict__ b1, const float* __restrict__ b2,
                      float* __restrict__ qc) {
    int b = blockIdx.x, tid = threadIdx.x;
#pragma unroll
    for (int i = 0; i < 4; ++i) {
        int u = tid + i * 256;
        float acc = 0.f;
        for (int h = 0; h < H_; ++h)
            acc += q[b * H_ + h] * W2[(size_t)h * U_ + u];
        qc[b * U_ + u] = acc + b1[u] + b2[u];
    }
}

// ---------------------------------------------------------------------------
// Kernel 3: fused  score[b,s] = V . tanh(values_row @ W1 + qc[b]) + bV
// WG = 256 threads (8 waves), owns 128 rows.  A tile staged in LDS as bf16 in
// FRAGMENT-MAJOR order: fragment (kb, m) is a contiguous 1 KB block at
// lds[(kb*8 + m)*512 ..), each lane owning 16 elems (32 B) at lane*16.
// kb loop is a hand-written period-2 software pipeline: loop-carried B frag b0
// is redefined only after its last use, so the allocator never rotates acc or
// operand registers (no v_mov copies, no extra WMMA hazard NOPs).
// ---------------------------------------------------------------------------
__launch_bounds__(256, 1)
__global__ void fused_score(const float* __restrict__ values,
                            const __bf16* __restrict__ Wb,
                            const float* __restrict__ qc,
                            const float* __restrict__ V,
                            const float* __restrict__ bV,
                            float* __restrict__ scores) {
    extern __shared__ __bf16 lds[];          // TILE_M * 1024 bf16 = 256 KB
    __shared__ float sred[8][TILE_M];        // per-wave row-score accumulators

    const int tid  = threadIdx.x;
    const int wave = tid >> 5;
    const int lane = tid & 31;
    const int half = (lane < 16) ? 0 : 1;
    const int ln   = lane & 15;
    const int row0 = blockIdx.x * TILE_M;    // tiles never straddle a batch (2048%128==0)
    const int b    = row0 >> 11;

    for (int i = tid; i < 8 * TILE_M; i += 256) (&sred[0][0])[i] = 0.f;

    // ---- stage A tile: 128 rows x 1024 f32 -> bf16 LDS, fragment-major ----
    // A-fragment element map: row = m*16 + ln2, K = kb*32 + ko with
    // half = ko[4], j = ko&15; lane slot = half*16 + ln2.
    const float4* src = (const float4*)(values + (size_t)row0 * H_);
    for (int idx = tid; idx < TILE_M * 256; idx += 256) {
        int r  = idx >> 8;                    // row in tile
        int c4 = idx & 255;                   // float4 index within row
        float4 v = src[(size_t)r * 256 + c4];
        int m   = r >> 4, ln2 = r & 15;
        int k0  = c4 << 2;
        int kb  = k0 >> 5, ko = k0 & 31;
        int hf  = ko >> 4;
        int j   = ko & 15;                       // float4 -> 4 contiguous j slots
        v4bf p = { (__bf16)v.x, (__bf16)v.y, (__bf16)v.z, (__bf16)v.w };
        *(v4bf*)(lds + ((kb * 8 + m) * 32 + hf * 16 + ln2) * 16 + j) = p;
    }
    __syncthreads();

    const __bf16* lanebase = lds + lane * 16;   // + (kb*8+m)*512 elems per fragment

    // ---- per-wave GEMM + tanh + dot(V) ----
    for (int t = 0; t < 8; ++t) {
        const int nt  = wave * 8 + t;
        const int u   = nt * 16 + ln;
        const float qcv = qc[b * U_ + u];
        const float Vu  = V[u];

        v8f acc[8] = {};                                  // 8 M-subtiles x 16x16 f32
        const __bf16* wp = Wb + ((size_t)nt * 32 + lane) * 16;

        v16bf b0 = *(const v16bf*)(wp);                   // B frag for kb=0
        const __bf16* abase = lanebase;

#pragma unroll 1
        for (int kb = 0; kb < 32; kb += 2) {
            // B frag for kb+1 (iteration-local)
            v16bf b1 = *(const v16bf*)(wp + (size_t)(kb + 1) * 32768);
            __builtin_prefetch((const void*)(wp + (size_t)((kb + 4) & 31) * 32768), 0, 3);

            // ---- step kb: 8 A frags (DS-immediate offsets), 8 WMMAs on b0 ----
            v16bf af[8];
#pragma unroll
            for (int m = 0; m < 8; ++m)
                af[m] = *(const v16bf*)(abase + m * 512);
#pragma unroll
            for (int m = 0; m < 8; ++m)
                acc[m] = __builtin_amdgcn_wmma_f32_16x16x32_bf16(
                    false, af[m], false, b0, (short)0, acc[m], false, false);

            // b0's last use is above -> redefine for kb+2 (wraps harmlessly at end)
            b0 = *(const v16bf*)(wp + (size_t)((kb + 2) & 31) * 32768);

            // ---- step kb+1: next 8 KB fragment block, 8 WMMAs on b1 ----
            v16bf ag[8];
#pragma unroll
            for (int m = 0; m < 8; ++m)
                ag[m] = *(const v16bf*)(abase + 4096 + m * 512);
#pragma unroll
            for (int m = 0; m < 8; ++m)
                acc[m] = __builtin_amdgcn_wmma_f32_16x16x32_bf16(
                    false, ag[m], false, b1, (short)0, acc[m], false, false);

            abase += 8192;                                // two 4 KB-elem blocks
        }

        // tanh, scale by V[u], reduce over the 16 N-lanes, accumulate row score
#pragma unroll
        for (int m = 0; m < 8; ++m) {
#pragma unroll
            for (int r = 0; r < 8; ++r) {
                float x = tanhf(acc[m][r] + qcv) * Vu;
                x += __shfl_xor(x, 1);
                x += __shfl_xor(x, 2);
                x += __shfl_xor(x, 4);
                x += __shfl_xor(x, 8);                    // halves stay separate
                if (ln == 0) sred[wave][m * 16 + half * 8 + r] += x;
            }
        }
    }
    __syncthreads();

    if (tid < TILE_M) {
        float s = 0.f;
#pragma unroll
        for (int w = 0; w < 8; ++w) s += sred[w][tid];
        scores[row0 + tid] = s + bV[0];
    }
}

// ---------------------------------------------------------------------------
// Kernel 4: softmax over S per batch -> attention weights (d_out[0 : B*S])
// ---------------------------------------------------------------------------
__global__ void softmax_k(const float* __restrict__ scores, float* __restrict__ w) {
    __shared__ float red[256];
    int b = blockIdx.x, tid = threadIdx.x;
    const float* s = scores + b * S_;
    float v[8], mx = -1e30f;
#pragma unroll
    for (int i = 0; i < 8; ++i) { v[i] = s[tid + i * 256]; mx = fmaxf(mx, v[i]); }
    red[tid] = mx; __syncthreads();
    for (int st = 128; st > 0; st >>= 1) {
        if (tid < st) red[tid] = fmaxf(red[tid], red[tid + st]);
        __syncthreads();
    }
    mx = red[0]; __syncthreads();
    float sum = 0.f;
#pragma unroll
    for (int i = 0; i < 8; ++i) { v[i] = __expf(v[i] - mx); sum += v[i]; }
    red[tid] = sum; __syncthreads();
    for (int st = 128; st > 0; st >>= 1) {
        if (tid < st) red[tid] += red[tid + st];
        __syncthreads();
    }
    float inv = 1.f / red[0];
#pragma unroll
    for (int i = 0; i < 8; ++i) w[b * S_ + tid + i * 256] = v[i] * inv;
}

// ---------------------------------------------------------------------------
// Kernel 5: context partials  part[b][sc][h] = sum_{s in chunk} w[b,s]*values[b,s,h]
// ---------------------------------------------------------------------------
__global__ void ctx_partial(const float* __restrict__ values,
                            const float* __restrict__ w,
                            float* __restrict__ part) {
    int b = blockIdx.x, hc = blockIdx.y, sc = blockIdx.z;
    int h = hc * 256 + threadIdx.x;
    const float* vp = values + ((size_t)b * S_ + sc * 512) * H_ + h;
    const float* wp = w + b * S_ + sc * 512;
    float acc = 0.f;
    for (int s = 0; s < 512; ++s)
        acc += wp[s] * vp[(size_t)s * H_];
    part[((size_t)b * 4 + sc) * H_ + h] = acc;
}

__global__ void ctx_reduce(const float* __restrict__ part, float* __restrict__ ctx) {
    int i = blockIdx.x * 256 + threadIdx.x;   // b*1024 + h
    int b = i >> 10, h = i & 1023;
    float s = 0.f;
#pragma unroll
    for (int c = 0; c < 4; ++c) s += part[((size_t)b * 4 + c) * H_ + h];
    ctx[i] = s;
}

// ---------------------------------------------------------------------------
extern "C" void kernel_launch(void* const* d_in, const int* in_sizes, int n_in,
                              void* d_out, int out_size, void* d_ws, size_t ws_size,
                              hipStream_t stream) {
    const float* query  = (const float*)d_in[0];
    const float* values = (const float*)d_in[1];
    const float* W1     = (const float*)d_in[2];
    const float* b1     = (const float*)d_in[3];
    const float* W2     = (const float*)d_in[4];
    const float* b2     = (const float*)d_in[5];
    const float* V      = (const float*)d_in[6];
    const float* bV     = (const float*)d_in[7];

    char* ws = (char*)d_ws;
    float*  qc     = (float*)(ws);                         // 128 KB
    __bf16* Wb     = (__bf16*)(ws + 131072);               // 2 MB
    float*  scores = (float*)(ws + 131072 + 2097152);      // 256 KB
    float*  part   = (float*)(ws + 131072 + 2097152 + 262144); // 512 KB

    float* weights = (float*)d_out;          // [B,S,1] flat
    float* ctx     = weights + B_ * S_;      // [B,H]

    swizzle_w1<<<256, 256, 0, stream>>>(W1, Wb);
    qproj<<<B_, 256, 0, stream>>>(query, W2, b1, b2, qc);

    size_t lds_bytes = (size_t)TILE_M * H_ * sizeof(__bf16);   // 262144 B
    fused_score<<<(B_ * S_) / TILE_M, 256, lds_bytes, stream>>>(
        values, Wb, qc, V, bV, scores);

    softmax_k<<<B_, 256, 0, stream>>>(scores, weights);
    ctx_partial<<<dim3(B_, 4, 4), 256, 0, stream>>>(values, weights, part);
    ctx_reduce<<<(B_ * H_) / 256, 256, 0, stream>>>(part, ctx);
}
